// NeuralGraphHidden_28965259444493
// MI455X (gfx1250) — matrix operand order repaired and checked by reading, not compiled, not run on hardware
//
#include <hip/hip_runtime.h>

// Problem dims (fixed by the reference)
#define B_ 256
#define A_ 128
#define F_ 512
#define D_ 6
#define C_ 512
#define BA_ (B_ * A_)          // 32768 rows total
#define ASTRIDE_B 1040         // LDS row stride: 1024B row + 16B TDM pad (bank-conflict free)

typedef __attribute__((ext_vector_type(16))) __bf16 v16bf;
typedef __attribute__((ext_vector_type(8)))  __bf16 v8bf;
typedef __attribute__((ext_vector_type(8)))  float  v8f;
typedef __attribute__((ext_vector_type(4)))  unsigned int u32x4;
typedef __attribute__((ext_vector_type(8)))  unsigned int u32x8;

__device__ __forceinline__ unsigned short f2bf(float f) {
  unsigned u = __float_as_uint(f);
  unsigned r = 0x7FFFu + ((u >> 16) & 1u);   // round-to-nearest-even
  return (unsigned short)((u + r) >> 16);
}

__device__ __forceinline__ unsigned rfl(unsigned v) {
  return (unsigned)__builtin_amdgcn_readfirstlane((int)v);
}

// ---- zero the per-degree counters (graph-capture safe) ----
__global__ void k_zero(int* __restrict__ p) { p[threadIdx.x] = 0; }

// ---- kernel 1: neighbour gather + self add, fp32 -> bf16, degree bucketing ----
__global__ __launch_bounds__(128)
void k_gather(const float* __restrict__ atoms, const int* __restrict__ bonds,
              unsigned short* __restrict__ summed, int* __restrict__ counts,
              int* __restrict__ rowlist) {
  const int row = blockIdx.x;          // 0..BA_-1
  const int b   = row / A_;
  const int tid = threadIdx.x;         // 0..127, 4 floats each
  __shared__ int sb[D_];
  if (tid < D_) sb[tid] = bonds[row * D_ + tid];
  __syncthreads();

  const float4* self = (const float4*)(atoms + (size_t)row * F_);
  float4 acc = self[tid];
  int deg = 0;
#pragma unroll
  for (int d = 0; d < D_; ++d) {
    int idx = sb[d];
    if (idx >= 0) {
      ++deg;
      const float4* nb = (const float4*)(atoms + (size_t)(b * A_ + idx) * F_);
      float4 v = nb[tid];
      acc.x += v.x; acc.y += v.y; acc.z += v.z; acc.w += v.w;
    }
  }
  uint2 pk;
  pk.x = (unsigned)f2bf(acc.x) | ((unsigned)f2bf(acc.y) << 16);
  pk.y = (unsigned)f2bf(acc.z) | ((unsigned)f2bf(acc.w) << 16);
  *(uint2*)(summed + (size_t)row * F_ + tid * 4) = pk;

  if (tid == 0) {
    int slot = atomicAdd(&counts[deg], 1);
    rowlist[deg * BA_ + slot] = row;
  }
}

// ---- kernel 2: W [D,F,C] fp32 -> Wt [D,C,F] bf16 (K-contiguous per column) ----
__global__ __launch_bounds__(256)
void k_wt(const float* __restrict__ W, unsigned short* __restrict__ Wt) {
  int i = blockIdx.x * 256 + threadIdx.x;          // over D_*C_*F_
  if (i >= D_ * C_ * F_) return;
  int f = i % F_;
  int t = i / F_;
  int c = t % C_;
  int d = t / C_;
  Wt[i] = f2bf(W[((size_t)d * F_ + f) * C_ + c]);
}

// ---- kernel 3: grouped GEMM by degree; TDM gather-mode stages the 16-row A tile
//      into LDS, bf16 WMMA computes 16 rows x 512 cols per block (8 waves). ----
__global__ __launch_bounds__(256)
void k_gemm(const __bf16* __restrict__ summed, const __bf16* __restrict__ Wt,
            const float* __restrict__ bias, const int* __restrict__ counts,
            const int* __restrict__ rowlist, float* __restrict__ out) {
  const int d     = blockIdx.y;
  const int count = counts[d];
  const int tile  = blockIdx.x;
  if (tile * 16 >= count) return;

  __shared__ int sRows[16];
  __shared__ __align__(16) unsigned char sA[16 * ASTRIDE_B];  // 16 rows x (1024B + 16B pad)

  const int tid = threadIdx.x;
  if (tid < 16) {
    int idx = tile * 16 + tid;
    if (idx > count - 1) idx = count - 1;          // clamp; masked at store
    sRows[tid] = rowlist[d * BA_ + idx];
  }
  __syncthreads();

  // Wave 0 issues one TDM gather: 16 scattered bf16 rows (512 elems each) -> LDS.
  if (tid < 32) {
    const unsigned long long ga = (unsigned long long)(uintptr_t)summed;
    const unsigned ldsOff = (unsigned)(uintptr_t)(void*)sA;  // generic addr low 32 = LDS offset

    u32x4 g0;
    g0[0] = 0x80000001u;                                 // count=1 | gather_mode=1 (16-bit idx)
    g0[1] = ldsOff;                                      // lds_addr
    g0[2] = (unsigned)(ga & 0xFFFFFFFFu);                // global_addr[31:0]
    g0[3] = (unsigned)((ga >> 32) & 0x01FFFFFFu) | (2u << 30);  // addr[56:32] | type=2

    u32x8 g1;
    g1[0] = (1u << 16)            // data_size = 2 bytes
          | (1u << 20)            // pad_enable
          | (7u << 22)            // pad_interval: every 256 DWORDs (1024B)
          | (3u << 25);           // pad_amount: 4 DWORDs (16B)
    g1[1] = (unsigned)F_  << 16;  // tensor_dim0 = 512   (bits 63:48)
    g1[2] = (unsigned)BA_ << 16;  // tensor_dim1 = 32768 (bits 95:80)
    g1[3] = (unsigned)F_  << 16;  // tile_dim0  = 512    (bits 127:112)
    g1[4] = 16u;                  // tile_dim1  = #row indices = 16
    g1[5] = (unsigned)F_;         // tensor_dim0_stride = 512 elements
    g1[6] = 0u;
    g1[7] = 0u;

    // 16-bit row indices, packed 2/dword; values are wave-uniform (same LDS words
    // read by all lanes) -> readfirstlane to force them into SGPRs for the "s"
    // constraints (LDS reads otherwise land in VGPRs and break the TDM encoding).
    u32x4 g2, g3;
    g2[0] = rfl(((unsigned)sRows[0]  & 0xFFFFu) | (((unsigned)sRows[1]  & 0xFFFFu) << 16));
    g2[1] = rfl(((unsigned)sRows[2]  & 0xFFFFu) | (((unsigned)sRows[3]  & 0xFFFFu) << 16));
    g2[2] = rfl(((unsigned)sRows[4]  & 0xFFFFu) | (((unsigned)sRows[5]  & 0xFFFFu) << 16));
    g2[3] = rfl(((unsigned)sRows[6]  & 0xFFFFu) | (((unsigned)sRows[7]  & 0xFFFFu) << 16));
    g3[0] = rfl(((unsigned)sRows[8]  & 0xFFFFu) | (((unsigned)sRows[9]  & 0xFFFFu) << 16));
    g3[1] = rfl(((unsigned)sRows[10] & 0xFFFFu) | (((unsigned)sRows[11] & 0xFFFFu) << 16));
    g3[2] = rfl(((unsigned)sRows[12] & 0xFFFFu) | (((unsigned)sRows[13] & 0xFFFFu) << 16));
    g3[3] = rfl(((unsigned)sRows[14] & 0xFFFFu) | (((unsigned)sRows[15] & 0xFFFFu) << 16));

    asm volatile("tensor_load_to_lds %0, %1, %2, %3"
                 :: "s"(g0), "s"(g1), "s"(g2), "s"(g3)
                 : "memory");
    __builtin_amdgcn_s_wait_tensorcnt(0);
  }
  __syncthreads();

  const int wave = tid >> 5;        // 8 waves; 64 output cols per wave
  const int lane = tid & 31;
  const int half = lane >> 4;       // K-half selector per ISA VGPR layouts
  const int r    = lane & 15;       // M (for A) / N (for B,C,D) within tile
  const int colBase = wave * 64;

  const __bf16* WtD = Wt + (size_t)d * (size_t)C_ * F_;
  // B layout (32x16 bf16): lanes 0-15 hold K=0..15, lanes 16-31 hold K=16..31,
  // column = lane&15 -> per-lane 32B contiguous at col*F + 16*half.
  const __bf16* bCol0 = WtD + (size_t)(colBase +  0 + r) * F_ + 16 * half;
  const __bf16* bCol1 = WtD + (size_t)(colBase + 16 + r) * F_ + 16 * half;
  const __bf16* bCol2 = WtD + (size_t)(colBase + 32 + r) * F_ + 16 * half;
  const __bf16* bCol3 = WtD + (size_t)(colBase + 48 + r) * F_ + 16 * half;

  // A fragment source in LDS: row r, elements kk+16h..+15
  const unsigned char* aBase = sA + (unsigned)r * ASTRIDE_B + 32 * half;

  v8f acc0 = {}, acc1 = {}, acc2 = {}, acc3 = {};

  for (int kk = 0; kk < F_; kk += 32) {
    v8bf alo = *(const v8bf*)(aBase + kk * 2);
    v8bf ahi = *(const v8bf*)(aBase + kk * 2 + 16);
    v16bf a = __builtin_shufflevector(alo, ahi,
        0, 1, 2, 3, 4, 5, 6, 7, 8, 9, 10, 11, 12, 13, 14, 15);
    v16bf b0 = *(const v16bf*)(bCol0 + kk);
    v16bf b1 = *(const v16bf*)(bCol1 + kk);
    v16bf b2 = *(const v16bf*)(bCol2 + kk);
    v16bf b3 = *(const v16bf*)(bCol3 + kk);
    acc0 = __builtin_amdgcn_wmma_f32_16x16x32_bf16(false, a, false, b0, (short)0, acc0, false, false);
    acc1 = __builtin_amdgcn_wmma_f32_16x16x32_bf16(false, a, false, b1, (short)0, acc1, false, false);
    acc2 = __builtin_amdgcn_wmma_f32_16x16x32_bf16(false, a, false, b2, (short)0, acc2, false, false);
    acc3 = __builtin_amdgcn_wmma_f32_16x16x32_bf16(false, a, false, b3, (short)0, acc3, false, false);
  }

  // Epilogue: C/D layout -> element i at lane L is (M = i + 8*(L>>4), N = L&15)
  const int mBase = tile * 16;
#pragma unroll
  for (int t = 0; t < 4; ++t) {
    const v8f av = (t == 0) ? acc0 : (t == 1) ? acc1 : (t == 2) ? acc2 : acc3;
    const int col = colBase + t * 16 + r;
    const float bv = bias[d * C_ + col];
#pragma unroll
    for (int i = 0; i < 8; ++i) {
      const int m = i + 8 * half;
      if (mBase + m < count)   // streaming, write-once output: bypass-friendly NT store
        __builtin_nontemporal_store(av[i] + bv, &out[(size_t)sRows[m] * C_ + col]);
    }
  }
}

extern "C" void kernel_launch(void* const* d_in, const int* in_sizes, int n_in,
                              void* d_out, int out_size, void* d_ws, size_t ws_size,
                              hipStream_t stream) {
  (void)in_sizes; (void)n_in; (void)out_size; (void)ws_size;
  const float* atoms = (const float*)d_in[0];
  const int*   bonds = (const int*)d_in[1];
  const float* W     = (const float*)d_in[2];
  const float* bias  = (const float*)d_in[3];
  float* out = (float*)d_out;

  // Workspace layout (~35.75 MiB):
  //   [0)            summed bf16 : BA_*F_*2   = 32 MiB
  //   [+32MiB)       Wt bf16     : D_*C_*F_*2 =  3 MiB
  //   [+35MiB)       counts      : 64 ints
  //   [+35MiB+256B)  rowlist     : D_*BA_ ints = 768 KiB
  char* ws = (char*)d_ws;
  unsigned short* summed = (unsigned short*)ws;
  unsigned short* Wt     = (unsigned short*)(ws + (size_t)BA_ * F_ * 2);
  int* counts  = (int*)(ws + (size_t)BA_ * F_ * 2 + (size_t)D_ * C_ * F_ * 2);
  int* rowlist = counts + 64;

  k_zero<<<1, 64, 0, stream>>>(counts);
  k_wt<<<(D_ * C_ * F_ + 255) / 256, 256, 0, stream>>>(W, Wt);
  k_gather<<<BA_, 128, 0, stream>>>(atoms, bonds, summed, counts, rowlist);
  k_gemm<<<dim3(BA_ / 16, D_), 256, 0, stream>>>(
      (const __bf16*)summed, (const __bf16*)Wt, bias, counts, rowlist, out);
}
